// Distillation_3977139716729
// MI455X (gfx1250) — matrix operand order repaired and checked by reading, not compiled, not run on hardware
//
#include <hip/hip_runtime.h>
#include <hip/hip_bf16.h>
#include <math.h>

typedef _Float16 v16h __attribute__((ext_vector_type(16)));
typedef float    v8f  __attribute__((ext_vector_type(8)));

// Fragment mapping (wave32, 16-bit WMMA):
//  A (16x32, row-major M x K): lane l -> M = l&15; halves e: k = kb + (e<8 ? e : 8+e),
//    kb = (l>=16)*8  (covers k in [kb,kb+8) U [kb+16,kb+24))
//  B (32x16, K x N): lane l -> N = l&15; halves e: k = (l>=16)*8 + (e<8 ? e : 8+e)
//  C/D f32: VGPR r -> M = r + 8*(l>=16), N = l&15

#define CVT4A(dst, i, f4) \
  dst[i] = (_Float16)(f4).x; dst[i+1] = (_Float16)(f4).y; \
  dst[i+2] = (_Float16)(f4).z; dst[i+3] = (_Float16)(f4).w;

// A fragment via LDS im2col offset table (kb already includes k0 + lane base).
// f32 source: gather + convert.
__device__ __forceinline__ v16h load_a_frag(const int* s_off,
                                            const float* __restrict__ xbase,
                                            int kb)
{
  int4 o0 = *(const int4*)&s_off[kb];
  int4 o1 = *(const int4*)&s_off[kb + 4];
  int4 o2 = *(const int4*)&s_off[kb + 16];
  int4 o3 = *(const int4*)&s_off[kb + 20];
  v16h a;
  a[0]  = (_Float16)xbase[o0.x]; a[1]  = (_Float16)xbase[o0.y];
  a[2]  = (_Float16)xbase[o0.z]; a[3]  = (_Float16)xbase[o0.w];
  a[4]  = (_Float16)xbase[o1.x]; a[5]  = (_Float16)xbase[o1.y];
  a[6]  = (_Float16)xbase[o1.z]; a[7]  = (_Float16)xbase[o1.w];
  a[8]  = (_Float16)xbase[o2.x]; a[9]  = (_Float16)xbase[o2.y];
  a[10] = (_Float16)xbase[o2.z]; a[11] = (_Float16)xbase[o2.w];
  a[12] = (_Float16)xbase[o3.x]; a[13] = (_Float16)xbase[o3.y];
  a[14] = (_Float16)xbase[o3.z]; a[15] = (_Float16)xbase[o3.w];
  return a;
}

// f16 source: direct d16 gather, no conversion VALU in the hot loop.
__device__ __forceinline__ v16h load_a_frag(const int* s_off,
                                            const _Float16* __restrict__ xbase,
                                            int kb)
{
  int4 o0 = *(const int4*)&s_off[kb];
  int4 o1 = *(const int4*)&s_off[kb + 4];
  int4 o2 = *(const int4*)&s_off[kb + 16];
  int4 o3 = *(const int4*)&s_off[kb + 20];
  v16h a;
  a[0]  = xbase[o0.x]; a[1]  = xbase[o0.y];
  a[2]  = xbase[o0.z]; a[3]  = xbase[o0.w];
  a[4]  = xbase[o1.x]; a[5]  = xbase[o1.y];
  a[6]  = xbase[o1.z]; a[7]  = xbase[o1.w];
  a[8]  = xbase[o2.x]; a[9]  = xbase[o2.y];
  a[10] = xbase[o2.z]; a[11] = xbase[o2.w];
  a[12] = xbase[o3.x]; a[13] = xbase[o3.y];
  a[14] = xbase[o3.z]; a[15] = xbase[o3.w];
  return a;
}

// B fragment: k in [kb,kb+8) U [kb+16,kb+24) from one f32 weight row (OIHW), 4x b128 loads.
__device__ __forceinline__ v16h load_b_frag(const float* __restrict__ wrow,
                                            int kb)
{
  float4 b0 = *(const float4*)&wrow[kb];
  float4 b1 = *(const float4*)&wrow[kb + 4];
  float4 b2 = *(const float4*)&wrow[kb + 16];
  float4 b3 = *(const float4*)&wrow[kb + 20];
  v16h b;
  CVT4A(b, 0, b0) CVT4A(b, 4, b1) CVT4A(b, 8, b2) CVT4A(b, 12, b3)
  return b;
}

// ---------------------------------------------------------------------------
// WMMA implicit-GEMM convolution (VALID, NCHW / OIHW), register-blocked over
// OC: one wave computes a 16 x (NT*16) output strip with NT accumulators, so
// the scattered A gather is amortized over NT WMMAs. K stepped by 64 (all
// conv K here are multiples of 64) -> two dependent WMMAs per acc per step.
// TIn/TOut select f32 or f16 activations (intermediate convs run f16->f16).
// ---------------------------------------------------------------------------
template <int NT, class TIn, class TOut>
__global__ __launch_bounds__(256) void conv_wmma_kernel(
    const TIn* __restrict__ X, const float* __restrict__ W,
    const float* __restrict__ bias, TOut* __restrict__ Y,
    int Bsz, int IC, int IH, int IW, int OH, int OW,
    int KH, int KW, int stride, int act)
{
  extern __shared__ int s_off[];            // K ints: im2col input offsets
  const int OC  = NT * 16;
  int OHW = OH * OW;
  int KHW = KH * KW;
  int K   = IC * KHW;
  for (int k = threadIdx.x; k < K; k += blockDim.x) {
    int ic = k / KHW; int r = k - ic * KHW;
    int ky = r / KW;  int kx = r - ky * KW;
    s_off[k] = (ic * IH + ky) * IW + kx;
  }
  __syncthreads();

  int wave = blockIdx.x * (blockDim.x >> 5) + (threadIdx.x >> 5);
  int M = Bsz * OHW;
  if (wave >= (M >> 4)) return;             // wave-uniform
  int mt = wave;                            // one M-tile per wave
  int lane = threadIdx.x & 31, half = lane >> 4, l15 = lane & 15;

  int row  = mt * 16 + l15;
  int bimg = row / OHW;
  int r2   = row - bimg * OHW;
  int oy   = r2 / OW, ox = r2 - oy * OW;
  const TIn* xbase =
      X + ((size_t)bimg * IC * IH + (size_t)oy * stride) * IW + ox * stride;

  int kbA = half * 8;
  int kbB = half * 8;
  const float* wrow[NT];
#pragma unroll
  for (int nt = 0; nt < NT; ++nt)
    wrow[nt] = W + (size_t)(nt * 16 + l15) * K;

  v8f acc[NT] = {};
  for (int k0 = 0; k0 < K; k0 += 64) {
    if (k0 + 64 < K)
      __builtin_prefetch(&wrow[0][k0 + 64 + kbB], 0, 3);
    v16h aLo = load_a_frag(s_off, xbase, k0 + kbA);
    v16h aHi = load_a_frag(s_off, xbase, k0 + 32 + kbA);
#pragma unroll
    for (int nt = 0; nt < NT; ++nt) {
      v16h bLo = load_b_frag(wrow[nt], k0 + kbB);
      v16h bHi = load_b_frag(wrow[nt], k0 + 32 + kbB);
      acc[nt] = __builtin_amdgcn_wmma_f32_16x16x32_f16(false, aLo, false, bLo,
                                                       (short)0, acc[nt], false, false);
      acc[nt] = __builtin_amdgcn_wmma_f32_16x16x32_f16(false, aHi, false, bHi,
                                                       (short)0, acc[nt], false, false);
    }
  }
#pragma unroll
  for (int nt = 0; nt < NT; ++nt) {
    int oc = nt * 16 + l15;
    float bv = bias[oc];
#pragma unroll
    for (int r = 0; r < 8; ++r) {
      int m = mt * 16 + r + half * 8;
      float v = acc[nt][r] + bv;
      if (act) v = fmaxf(v, 0.f);
      int b2i = m / OHW; int rr = m - b2i * OHW;   // NCHW output
      Y[((size_t)b2i * OC + oc) * OHW + rr] = (TOut)v;
    }
  }
}

// ---------------------------------------------------------------------------
// WMMA dense GEMM: C[M,N] = act(A[M,K] @ W[K,N] + bias[N])
// One wave per 16x16 tile; guard-free b128 main loop + branch-free ragged
// tail (value selects, clamped addresses). act: 0=none, 1=relu, 2=sigmoid
// ---------------------------------------------------------------------------
__global__ __launch_bounds__(256) void gemm_wmma_kernel(
    const float* __restrict__ A, const float* __restrict__ W,
    const float* __restrict__ bias, float* __restrict__ C,
    int M, int N, int K, int act)
{
  const int wavesPerBlock = blockDim.x >> 5;
  int wave = blockIdx.x * wavesPerBlock + (threadIdx.x >> 5);
  int Mt = M >> 4;
  int Nt = (N + 15) >> 4;
  if (wave >= Mt * Nt) return;              // wave-uniform
  int mt = wave / Nt, nt = wave % Nt;
  int lane = threadIdx.x & 31;
  int half = lane >> 4;
  int l15  = lane & 15;
  int rowA = mt * 16 + l15;
  int kbA  = half * 8;
  int colB = nt * 16 + l15;
  int colC = (colB < N) ? colB : 0;         // clamped (loads stay in-bounds)
  int kbB  = half * 8;
  const float* arow = A + (size_t)rowA * K;

  v8f acc = {};
  int Kfull = K & ~31;
  for (int k0 = 0; k0 < Kfull; k0 += 32) {
    float4 a0 = *(const float4*)&arow[k0 + kbA];
    float4 a1 = *(const float4*)&arow[k0 + kbA + 4];
    float4 a2 = *(const float4*)&arow[k0 + kbA + 16];
    float4 a3 = *(const float4*)&arow[k0 + kbA + 20];
    v16h a, b;
    CVT4A(a, 0, a0) CVT4A(a, 4, a1) CVT4A(a, 8, a2) CVT4A(a, 12, a3)
#pragma unroll
    for (int e = 0; e < 16; ++e) {          // W column colB, stride N
      float vb = W[(size_t)(k0 + kbB + ((e < 8) ? e : 8 + e)) * N + colC];
      b[e] = (_Float16)((colB < N) ? vb : 0.f);
    }
    acc = __builtin_amdgcn_wmma_f32_16x16x32_f16(false, a, false, b,
                                                 (short)0, acc, false, false);
  }
  if (Kfull < K) {                          // ragged-K tail (wave-uniform)
    v16h a, b;
#pragma unroll
    for (int e = 0; e < 16; ++e) {
      int k  = Kfull + kbA + ((e < 8) ? e : 8 + e);
      int kc = (k < K) ? k : 0;
      float va = arow[kc];                  // unconditional load, value select
      a[e] = (_Float16)((k < K) ? va : 0.f);
    }
#pragma unroll
    for (int e = 0; e < 16; ++e) {
      int k  = Kfull + kbB + ((e < 8) ? e : 8 + e);
      int kc = (k < K) ? k : 0;
      float vb = W[(size_t)kc * N + colC];
      b[e] = (_Float16)((k < K && colB < N) ? vb : 0.f);
    }
    acc = __builtin_amdgcn_wmma_f32_16x16x32_f16(false, a, false, b,
                                                 (short)0, acc, false, false);
  }
  if (colB < N) {
    float bv = bias[colB];
#pragma unroll
    for (int r = 0; r < 8; ++r) {
      int m = mt * 16 + r + half * 8;
      float v = acc[r] + bv;
      if (act == 1)      v = fmaxf(v, 0.f);
      else if (act == 2) v = 1.f / (1.f + __expf(-v));
      C[(size_t)m * N + colB] = v;
    }
  }
}

// ---------------------------------------------------------------------------
// Spatial mean pool: one wave per (b,c) channel, shfl_xor reduction.
// ---------------------------------------------------------------------------
__global__ void meanpool_kernel(const float* __restrict__ x,
                                float* __restrict__ y, int BC, int S)
{
  int wave = blockIdx.x * (blockDim.x >> 5) + (threadIdx.x >> 5);
  if (wave >= BC) return;
  int lane = threadIdx.x & 31;
  const float* p = x + (size_t)wave * S;
  float s = 0.f;
  for (int j = lane; j < S; j += 32) s += p[j];
#pragma unroll
  for (int o = 16; o > 0; o >>= 1) s += __shfl_xor(s, o, 32);
  if (lane == 0) y[wave] = s / (float)S;
}

// ---------------------------------------------------------------------------
// Concat along dim 1: y[b, :D] = a[b], y[b, D:2D] = c[b]
// ---------------------------------------------------------------------------
__global__ void concat_kernel(const float* __restrict__ a,
                              const float* __restrict__ c,
                              float* __restrict__ y, int B, int D)
{
  int i = blockIdx.x * blockDim.x + threadIdx.x;
  if (i >= B * 2 * D) return;
  int b = i / (2 * D), j = i - b * 2 * D;
  y[i] = (j < D) ? a[b * D + j] : c[b * D + (j - D)];
}

// ---------------------------------------------------------------------------
// Head: playbook distances, distill row loss, argmin quantize, mixture, a_mse.
// One 128-thread block per batch row (N_W == 128).
// ---------------------------------------------------------------------------
__global__ void head_kernel(const float* __restrict__ w0,
                            const float* __restrict__ playbook,
                            const float* __restrict__ l0,
                            const float* __restrict__ subH3,
                            const float* __restrict__ a_t,
                            float* __restrict__ rowDistill,
                            float* __restrict__ out)
{
  int b = blockIdx.x, tid = threadIdx.x;   // tid = playbook entry n (0..127)
  __shared__ float sw0[16];
  __shared__ float red[128];
  __shared__ int   ridx[128];
  if (tid < 16) sw0[tid] = w0[b * 16 + tid];
  __syncthreads();

  // d[n] = || w0[b] - sigmoid(pb[n]) ||^2
  float dn = 0.f;
  for (int j = 0; j < 16; ++j) {
    float p = 1.f / (1.f + __expf(-playbook[tid * 16 + j]));
    float diff = sw0[j] - p;
    dn += diff * diff;
  }

  // argmin d (first index wins ties)
  red[tid] = dn; ridx[tid] = tid;
  __syncthreads();
  for (int s = 64; s > 0; s >>= 1) {
    if (tid < s) {
      float v2 = red[tid + s]; int i2 = ridx[tid + s];
      if (v2 < red[tid] || (v2 == red[tid] && i2 < ridx[tid])) {
        red[tid] = v2; ridx[tid] = i2;
      }
    }
    __syncthreads();
  }
  float dmin = red[0]; int qidx = ridx[0];
  __syncthreads();

  // log_softmax(-d): max(-d) = -dmin
  red[tid] = __expf(dmin - dn);
  __syncthreads();
  for (int s = 64; s > 0; s >>= 1) { if (tid < s) red[tid] += red[tid + s]; __syncthreads(); }
  float lse = __logf(red[0]);
  __syncthreads();
  float ppred = (dmin - dn) - lse;

  // p_true = softmax(-l0/1e-5)
  float t = -l0[b * 128 + tid] * 1e5f;
  red[tid] = t; __syncthreads();
  for (int s = 64; s > 0; s >>= 1) { if (tid < s) red[tid] = fmaxf(red[tid], red[tid + s]); __syncthreads(); }
  float tmax = red[0]; __syncthreads();
  float e = __expf(t - tmax);
  red[tid] = e; __syncthreads();
  for (int s = 64; s > 0; s >>= 1) { if (tid < s) red[tid] += red[tid + s]; __syncthreads(); }
  float pt = e / red[0];
  __syncthreads();

  red[tid] = pt * ppred; __syncthreads();
  for (int s = 64; s > 0; s >>= 1) { if (tid < s) red[tid] += red[tid + s]; __syncthreads(); }

  if (tid == 0) {
    rowDistill[b] = red[0];
    // inverse-variance mixture with quantized playbook weights
    float amse = 0.f;
    for (int j = 0; j < 6; ++j) {
      float num = 0.f, den = 0.f;
      for (int s = 0; s < 16; ++s) {
        float wq  = 1.f / (1.f + __expf(-playbook[qidx * 16 + s]));
        float mu  = subH3[((size_t)s * 128 + b) * 12 + j];
        float pre = subH3[((size_t)s * 128 + b) * 12 + 6 + j];
        float sp  = (pre > 20.f) ? pre : log1pf(__expf(pre));  // softplus
        float sd  = sp + 0.001f;                               // std = softplus(pre) + 0.001
        float inv = wq / (sd + 1e-6f);
        num += inv * mu; den += inv;
      }
      float mean = num / (den + 1e-6f);
      float diff = a_t[b * 7 + j] - mean;
      amse += diff * diff;
    }
    out[1 + b] = amse * (1.f / 6.f);
  }
}

__global__ void finalize_kernel(const float* __restrict__ rowDistill,
                                float* __restrict__ out)
{
  __shared__ float red[128];
  int tid = threadIdx.x;
  red[tid] = rowDistill[tid];
  __syncthreads();
  for (int s = 64; s > 0; s >>= 1) { if (tid < s) red[tid] += red[tid + s]; __syncthreads(); }
  if (tid == 0) { out[0] = -red[0] / 128.f; out[129] = 0.f; }
}

// ---------------------------------------------------------------------------
// Host orchestration
// ---------------------------------------------------------------------------
enum {
  IN_X0 = 0, IN_XT, IN_H0, IN_L0, IN_AT,
  IN_C1W, IN_C1B, IN_C2W, IN_C2B, IN_C3W, IN_C3B,
  IN_F1W, IN_F1B, IN_F2W, IN_F2B, IN_F3W, IN_F3B,
  IN_SKW1, IN_SKB1, IN_SKW2, IN_SKB2,
  IN_AGW1, IN_AGB1, IN_AGW2, IN_AGB2, IN_AGW3, IN_AGB3,
  IN_WNW1, IN_WNB1, IN_WNW2, IN_WNB2, IN_WNW3, IN_WNB3,
  IN_PB,
  IN_SC1W, IN_SC1B, IN_SC2W, IN_SC2B, IN_SC3W, IN_SC3B,
  IN_SF1W, IN_SF1B, IN_SF2W, IN_SF2B, IN_SF3W, IN_SF3B,
  IN_S1W, IN_S1B, IN_S2W, IN_S2B, IN_S3W, IN_S3B
};

extern "C" void kernel_launch(void* const* d_in, const int* in_sizes, int n_in,
                              void* d_out, int out_size, void* d_ws, size_t ws_size,
                              hipStream_t stream)
{
  (void)in_sizes; (void)n_in; (void)out_size; (void)ws_size;
  auto F = [&](int i) { return (const float*)d_in[i]; };

  const float* x0 = F(IN_X0);
  const float* xt = F(IN_XT);
  const float* h0 = F(IN_H0);
  const float* l0 = F(IN_L0);
  const float* at = F(IN_AT);

  // workspace carve (floats)
  float* ws = (float*)d_ws;
  size_t off = 0;
  auto alloc = [&](size_t n) { float* p = ws + off; off += n; return p; };
  _Float16* act1h = (_Float16*)alloc(128ull * 32 * 31 * 31 / 2 + 8);  // conv1 out (f16)
  _Float16* act2h = (_Float16*)alloc(128ull * 64 * 14 * 14 / 2 + 8);  // conv2 out (f16)
  float* act3   = alloc(128ull * 64 * 12 * 12);   // conv3 out (f32)
  float* poolb  = alloc(128ull * 64);
  float* t400   = alloc(128ull * 400);
  float* t300   = alloc(128ull * 300);
  float* xi     = alloc(128ull * 64);
  float* h256   = alloc(128ull * 256);
  float* hsb    = alloc(128ull * 64);
  float* xcat   = alloc(128ull * 128);
  float* a256a  = alloc(128ull * 256);
  float* a256b  = alloc(128ull * 256);
  float* fwbuf  = alloc(128ull * 64);
  float* w0buf  = alloc(128ull * 16);
  float* fbuf   = alloc(128ull * 64);
  float* t64a   = alloc(128ull * 64);
  float* t64b   = alloc(128ull * 64);
  float* subH3  = alloc(16ull * 128 * 12);
  float* rowD   = alloc(128);

  auto gemm = [&](const float* A, const float* W, const float* Bi, float* C,
                  int M, int N, int K, int act) {
    int tiles = (M / 16) * ((N + 15) / 16);
    gemm_wmma_kernel<<<dim3((tiles + 7) / 8), dim3(256), 0, stream>>>(
        A, W, Bi, C, M, N, K, act);
  };
  auto pool = [&](const float* X, float* Y, int BC, int S) {
    meanpool_kernel<<<dim3((BC + 7) / 8), dim3(256), 0, stream>>>(X, Y, BC, S);
  };
  // shared img2feat pipeline (stores f3 output to `feat`)
  auto img2feat = [&](const float* X, const float* c1w, const float* c1b,
                      const float* c2w, const float* c2b, const float* c3w,
                      const float* c3b, const float* f1w, const float* f1b,
                      const float* f2w, const float* f2b, const float* f3w,
                      const float* f3b, float* feat) {
    // conv1: 128x3x128x128 -> 128x32x31x31 (f16 out), M-tiles=7688
    conv_wmma_kernel<2, float, _Float16>
        <<<dim3(7688 / 8), dim3(256), 192 * sizeof(int), stream>>>(
        X, c1w, c1b, act1h, 128, 3, 128, 128, 31, 31, 8, 8, 4, 1);
    // conv2: -> 128x64x14x14 (f16 in/out), M-tiles=1568
    conv_wmma_kernel<4, _Float16, _Float16>
        <<<dim3(1568 / 8), dim3(256), 512 * sizeof(int), stream>>>(
        act1h, c2w, c2b, act2h, 128, 32, 31, 31, 14, 14, 4, 4, 2, 1);
    // conv3: -> 128x64x12x12 (f16 in, f32 out, no relu), M-tiles=1152
    conv_wmma_kernel<4, _Float16, float>
        <<<dim3(1152 / 8), dim3(256), 576 * sizeof(int), stream>>>(
        act2h, c3w, c3b, act3, 128, 64, 14, 14, 12, 12, 3, 3, 1, 0);
    pool(act3, poolb, 128 * 64, 144);
    gemm(poolb, f1w, f1b, t400, 128, 400, 64, 1);
    gemm(t400,  f2w, f2b, t300, 128, 300, 400, 1);
    gemm(t300,  f3w, f3b, feat, 128, 64, 300, 1);
  };

  // ---- fn path: xi = img2feat(x_0) ----
  img2feat(x0, F(IN_C1W), F(IN_C1B), F(IN_C2W), F(IN_C2B), F(IN_C3W), F(IN_C3B),
           F(IN_F1W), F(IN_F1B), F(IN_F2W), F(IN_F2B), F(IN_F3W), F(IN_F3B), xi);

  // ---- skill MLP ----
  gemm(h0,   F(IN_SKW1), F(IN_SKB1), h256, 128, 256, 16, 1);
  gemm(h256, F(IN_SKW2), F(IN_SKB2), hsb,  128, 64, 256, 1);

  // ---- concat + aggregator ----
  concat_kernel<<<dim3((128 * 128 + 255) / 256), dim3(256), 0, stream>>>(
      xi, hsb, xcat, 128, 64);
  gemm(xcat,  F(IN_AGW1), F(IN_AGB1), a256a, 128, 256, 128, 1);
  gemm(a256a, F(IN_AGW2), F(IN_AGB2), a256b, 128, 256, 256, 1);
  gemm(a256b, F(IN_AGW3), F(IN_AGB3), fwbuf, 128, 64, 256, 1);

  // ---- w-network (sigmoid output) ----
  gemm(fwbuf, F(IN_WNW1), F(IN_WNB1), t400,  128, 400, 64, 1);
  gemm(t400,  F(IN_WNW2), F(IN_WNB2), t300,  128, 300, 400, 1);
  gemm(t300,  F(IN_WNW3), F(IN_WNB3), w0buf, 128, 16, 300, 2);

  // ---- 16 sub-policies on x_t (reuse conv scratch; serialize host-side) ----
  for (int s = 0; s < 16; ++s) {
    img2feat(xt,
             F(IN_SC1W) + (size_t)s * 6144,   F(IN_SC1B) + (size_t)s * 32,
             F(IN_SC2W) + (size_t)s * 32768,  F(IN_SC2B) + (size_t)s * 64,
             F(IN_SC3W) + (size_t)s * 36864,  F(IN_SC3B) + (size_t)s * 64,
             F(IN_SF1W) + (size_t)s * 25600,  F(IN_SF1B) + (size_t)s * 400,
             F(IN_SF2W) + (size_t)s * 120000, F(IN_SF2B) + (size_t)s * 300,
             F(IN_SF3W) + (size_t)s * 19200,  F(IN_SF3B) + (size_t)s * 64,
             fbuf);
    gemm(fbuf, F(IN_S1W) + (size_t)s * 4096, F(IN_S1B) + (size_t)s * 64,
         t64a, 128, 64, 64, 1);
    gemm(t64a, F(IN_S2W) + (size_t)s * 4096, F(IN_S2B) + (size_t)s * 64,
         t64b, 128, 64, 64, 1);
    gemm(t64b, F(IN_S3W) + (size_t)s * 768,  F(IN_S3B) + (size_t)s * 12,
         subH3 + (size_t)s * 128 * 12, 128, 12, 64, 0);
  }

  // ---- head + reductions ----
  head_kernel<<<dim3(128), dim3(128), 0, stream>>>(
      w0buf, F(IN_PB), l0, subH3, at, rowD, (float*)d_out);
  finalize_kernel<<<dim3(1), dim3(128), 0, stream>>>(rowD, (float*)d_out);
}
